// cylinder_fea_33689723470419
// MI455X (gfx1250) — compile-verified, matrix operand order checked by reading, not run on hardware
//
#include <hip/hip_runtime.h>
#include <hip/hip_bf16.h>

typedef __attribute__((ext_vector_type(16))) _Float16 v16h;
typedef __attribute__((ext_vector_type(8)))  _Float16 v8h;
typedef __attribute__((ext_vector_type(8)))  float    v8f;

union AF { v16h v; v8h h[2]; };

// ---------------------------------------------------------------------------
// Pointers into workspace (folded biases, scales, WMMA-packed f16 weights)
// ---------------------------------------------------------------------------
struct WsPtrs {
  float* ori;                                   // V x 256 voxel max-pool of mlp_fea
  float *a0, *c0, *a1, *a2, *a3;                // BN fold scales
  float *b1, *b2, *b3, *b4;                     // folded biases
  float *bq, *bk, *bv, *bo, *bf1, *bf2, *bc1, *bc2;
  _Float16 *w1, *w2, *w3, *w4, *wq, *wk, *wv, *wo, *wf1, *wf2, *wc1, *wc2;
};

struct Params {
  const float *bn0g,*bn0b,*bn0m,*bn0v;
  const float *l1W,*l1b, *bn1g,*bn1b,*bn1m,*bn1v;
  const float *l2W,*l2b, *bn2g,*bn2b,*bn2m,*bn2v;
  const float *l3W,*l3b, *bn3g,*bn3b,*bn3m,*bn3v;
  const float *l4W,*l4b;
  const float *qW,*qb,*kW,*kb,*vW,*vb,*oW,*ob;
  const float *ln1g,*ln1b;
  const float *f1W,*f1b,*f2W,*f2b;
  const float *ln2g,*ln2b;
  const float *c1W,*c1b,*c2W,*c2b;
};

// ---------------------------------------------------------------------------
// Wave-level 16xN = (16xK) @ (KxN) GEMM on v_wmma_f32_16x16x32_f16.
// KT = K/32 is a compile-time constant; all A fragments are preloaded into
// registers once (2 ds_load_b128 per fragment) and reused across all NT
// output tiles.  B fragments stream from the pre-packed global weight image
// as two coalesced global_load_b128 per fragment (L2-resident).
// A-fragment layout (ISA 7.12.2, 16-bit A 16x32): lane L holds row M=L&15;
// halves 0..7 = K = kt*32 + (L&16?8:0) + 0..7, halves 8..15 = +16.
// D layout: lane L, vgpr r -> row = r + (L&16?8:0), col = L&15.
// ---------------------------------------------------------------------------
template<int KT>
__device__ __forceinline__ void wave_gemm(
    const _Float16* __restrict__ hA, int ldk,
    const _Float16* __restrict__ Wpk, const float* __restrict__ bias, int NT,
    _Float16* __restrict__ hOut, int ldo,
    float* __restrict__ fOut, int ldf,
    bool relu, int lane)
{
  const int m    = lane & 15;
  const int half = (lane & 16) ? 8 : 0;   // K-base offset and D row-base
  const int col  = lane & 15;

  // Preload all A fragments into registers (KT <= 16 -> <= 128 VGPRs).
  AF afr[KT];
#pragma unroll
  for (int kt = 0; kt < KT; ++kt) {
    const _Float16* ap = hA + m * ldk + kt * 32 + half;
    afr[kt].h[0] = *(const v8h*)(ap);        // ds_load_b128
    afr[kt].h[1] = *(const v8h*)(ap + 16);
  }

  for (int nt = 0; nt < NT; ++nt) {
    float bc = bias[nt * 16 + col];          // unconditional, hoisted over WMMAs
    v8f acc = {0.f,0.f,0.f,0.f,0.f,0.f,0.f,0.f};
    const _Float16* wp = Wpk + (size_t)(nt * KT) * 512 + lane * 16;
#pragma unroll
    for (int kt = 0; kt < KT; ++kt) {
      AF b;
      b.h[0] = *(const v8h*)(wp);            // global_load_b128
      b.h[1] = *(const v8h*)(wp + 8);
      __builtin_prefetch(wp + 512, 0, 3);    // WGP-scope prefetch of next frag
      acc = __builtin_amdgcn_wmma_f32_16x16x32_f16(false, afr[kt].v, false, b.v,
                                                   (short)0, acc, false, false);
      wp += 512;
    }
#pragma unroll
    for (int r = 0; r < 8; ++r) {
      float x = acc[r] + bc;
      if (relu) x = fmaxf(x, 0.f);
      int row = half + r;
      if (hOut) hOut[row * ldo + nt * 16 + col] = (_Float16)x;
      if (fOut) fOut[row * ldf + nt * 16 + col] = x;
    }
  }
}

// ---------------------------------------------------------------------------
// Fold BN parameters into linear weights/biases (tiny, one block).
// h = relu(bn(lin(bn0(x))))  ==>  h = relu(x @ W' + b')
// ---------------------------------------------------------------------------
__global__ void fold_kernel(Params P, WsPtrs W)
{
  int t = threadIdx.x;
  if (t < 3) {
    float a = P.bn0g[t] * rsqrtf(P.bn0v[t] + 1e-5f);
    W.a0[t] = a;
    W.c0[t] = P.bn0b[t] - P.bn0m[t] * a;
  }
  __syncthreads();
  if (t < 64) {
    float a = P.bn1g[t] * rsqrtf(P.bn1v[t] + 1e-5f);
    W.a1[t] = a;
    float s = P.l1b[t];
    for (int k = 0; k < 3; ++k) s += W.c0[k] * P.l1W[k * 64 + t];
    W.b1[t] = s * a + (P.bn1b[t] - P.bn1m[t] * a);
  }
  if (t < 128) {
    float a = P.bn2g[t] * rsqrtf(P.bn2v[t] + 1e-5f);
    W.a2[t] = a;
    W.b2[t] = P.l2b[t] * a + (P.bn2b[t] - P.bn2m[t] * a);
    W.bc2[t] = P.c2b[t];
  }
  if (t < 256) {
    float a = P.bn3g[t] * rsqrtf(P.bn3v[t] + 1e-5f);
    W.a3[t] = a;
    W.b3[t] = P.l3b[t] * a + (P.bn3b[t] - P.bn3m[t] * a);
    W.b4[t] = P.l4b[t];
    W.bq[t] = P.qb[t];  W.bk[t] = P.kb[t];
    W.bv[t] = P.vb[t];  W.bo[t] = P.ob[t];
    W.bf2[t] = P.f2b[t]; W.bc1[t] = P.c1b[t];
  }
  if (t < 512) W.bf1[t] = P.f1b[t];
}

// ---------------------------------------------------------------------------
// Pack a (K x N) f32 weight into WMMA B-fragment order as f16, applying
// optional row/col BN scales.  K is padded to KT*32 with zeros.
// ---------------------------------------------------------------------------
__global__ void pack_kernel(const float* __restrict__ Wsrc, int K, int N, int KT,
                            const float* __restrict__ rowS,
                            const float* __restrict__ colS,
                            _Float16* __restrict__ dst)
{
  int total = KT * (N >> 4) * 512;
  for (int e = blockIdx.x * blockDim.x + threadIdx.x; e < total;
       e += gridDim.x * blockDim.x) {
    int frag = e >> 9, r = e & 511, lane = r >> 4, h = r & 15;
    int nt = frag / KT, kt = frag % KT;
    int v = h >> 1;
    int klocal = ((v & 3) * 2) + (h & 1) + ((lane & 16) ? 8 : 0) + ((v >= 4) ? 16 : 0);
    int k = kt * 32 + klocal;
    int n = nt * 16 + (lane & 15);
    float val = 0.f;
    if (k < K) {
      val = Wsrc[(size_t)k * N + n];
      if (rowS) val *= rowS[k];
      if (colS) val *= colS[n];
    }
    dst[e] = (_Float16)val;
  }
}

// ---------------------------------------------------------------------------
// Init pooled buffers to -inf (segment_max identity).
// ---------------------------------------------------------------------------
__global__ void init_pool_kernel(float* ori, float* s1, float* s2, float* s3,
                                 float* pixp, long long total)
{
  long long i = (long long)blockIdx.x * blockDim.x + threadIdx.x;
  if (i >= total) return;
  float ninf = -__builtin_huge_valf();
  ori[i] = ninf; s1[i] = ninf; s2[i] = ninf; s3[i] = ninf; pixp[i] = ninf;
}

__global__ void copy_int_kernel(const int* unq, const int* unq_inv,
                                int* outUnq, int* outInv, int nUnq, int nInv)
{
  int i = blockIdx.x * blockDim.x + threadIdx.x;
  if (i < nUnq) outUnq[i] = unq[i];
  if (i < nInv) outInv[i] = unq_inv[i];
}

// ---------------------------------------------------------------------------
// Point MLP: one wave per 16-point tile, 4 waves per block.
// 3->64->128->256->256 fully on WMMA; layer-4 epilogue scatters with
// float atomic-max directly from D fragments into the voxel buffer.
// ---------------------------------------------------------------------------
__global__ void mlp_kernel(const float* __restrict__ pt,
                           const int* __restrict__ unq_inv,
                           WsPtrs W, int N)
{
  extern __shared__ __align__(16) char smem[];
  const int wave = threadIdx.x >> 5, lane = threadIdx.x & 31;
  _Float16* hA = (_Float16*)smem + (size_t)wave * (2 * 16 * 256);
  _Float16* hB = hA + 16 * 256;
  const int tile = blockIdx.x * 4 + wave;
  const int p0 = tile * 16;
  if (p0 >= N) return;

  int vid = 0;
  if (lane < 16) vid = unq_inv[p0 + lane];

  // Stage raw xyz as f16 A-tile (K padded 3 -> 32); BN0 folded into W1.
  for (int idx = lane; idx < 16 * 32; idx += 32) {
    int r = idx >> 5, c = idx & 31;
    hA[r * 256 + c] = (c < 3) ? (_Float16)pt[(size_t)(p0 + r) * 3 + c] : (_Float16)0.f;
  }

  wave_gemm<1>(hA, 256, W.w1, W.b1, 4,  hB, 256, nullptr, 0, true, lane);  // 3->64
  wave_gemm<2>(hB, 256, W.w2, W.b2, 8,  hA, 256, nullptr, 0, true, lane);  // 64->128
  wave_gemm<4>(hA, 256, W.w3, W.b3, 16, hB, 256, nullptr, 0, true, lane);  // 128->256

  // Layer 4 (256->256) with atomic-max scatter epilogue.
  {
    const int m = lane & 15, half = (lane & 16) ? 8 : 0, col = lane & 15;
    AF afr[8];
#pragma unroll
    for (int kt = 0; kt < 8; ++kt) {
      const _Float16* ap = hB + m * 256 + kt * 32 + half;
      afr[kt].h[0] = *(const v8h*)(ap);
      afr[kt].h[1] = *(const v8h*)(ap + 16);
    }
    for (int nt = 0; nt < 16; ++nt) {
      float bc = W.b4[nt * 16 + col];
      v8f acc = {0.f,0.f,0.f,0.f,0.f,0.f,0.f,0.f};
      const _Float16* wp = W.w4 + (size_t)(nt * 8) * 512 + lane * 16;
#pragma unroll
      for (int kt = 0; kt < 8; ++kt) {
        AF b;
        b.h[0] = *(const v8h*)(wp);
        b.h[1] = *(const v8h*)(wp + 8);
        __builtin_prefetch(wp + 512, 0, 3);
        acc = __builtin_amdgcn_wmma_f32_16x16x32_f16(false, afr[kt].v, false, b.v,
                                                     (short)0, acc, false, false);
        wp += 512;
      }
#pragma unroll
      for (int r = 0; r < 8; ++r) {
        int row = half + r;
        int vox = __shfl(vid, row, 32);
        float x = acc[r] + bc;
        unsafeAtomicMax(&W.ori[(size_t)vox * 256 + nt * 16 + col], x);  // global_atomic_max_num_f32
      }
    }
  }
}

// ---------------------------------------------------------------------------
// Segment-max pooling for segfea slices and pixfea.
// ---------------------------------------------------------------------------
__global__ void pool_kernel(const float* __restrict__ segfea,
                            const float* __restrict__ pixfea,
                            const int* __restrict__ unq_inv,
                            float* s1, float* s2, float* s3, float* pixp,
                            long long total)
{
  long long idx = (long long)blockIdx.x * blockDim.x + threadIdx.x;
  if (idx >= total) return;
  int i = (int)(idx >> 8), c = (int)(idx & 255);
  int v = unq_inv[i];
  size_t o = (size_t)v * 256 + c;
  size_t b = (size_t)i * 768;
  unsafeAtomicMax(&s1[o],   segfea[b + c]);
  unsafeAtomicMax(&s2[o],   segfea[b + 256 + c]);
  unsafeAtomicMax(&s3[o],   segfea[b + 512 + c]);
  unsafeAtomicMax(&pixp[o], pixfea[(size_t)i * 256 + c]);
}

// ---------------------------------------------------------------------------
// LayerNorm row statistics: 2 lanes per row, combine with shfl_xor(16).
// ---------------------------------------------------------------------------
__device__ __forceinline__ void ln_rows(const float* x, float* mu, float* rs, int lane)
{
  int r = lane & 15, part = lane >> 4;
  float s = 0.f, s2 = 0.f;
  int cb = part * 128;
  for (int c = 0; c < 128; ++c) { float v = x[r * 256 + cb + c]; s += v; s2 += v * v; }
  s  += __shfl_xor(s, 16, 32);
  s2 += __shfl_xor(s2, 16, 32);
  if (lane < 16) {
    float m = s * (1.f / 256.f);
    mu[r] = m;
    rs[r] = rsqrtf(s2 * (1.f / 256.f) - m * m + 1e-5f);
  }
}

// ---------------------------------------------------------------------------
// Per-voxel attention + FFN + fusion head; one wave handles 16 voxels so
// every GEMM is a dense 16xNxK WMMA problem.
// ---------------------------------------------------------------------------
__global__ void attn_kernel(const float* __restrict__ s1, const float* __restrict__ s2,
                            const float* __restrict__ s3, const float* __restrict__ pix,
                            const float* __restrict__ ln1g, const float* __restrict__ ln1b,
                            const float* __restrict__ ln2g, const float* __restrict__ ln2b,
                            WsPtrs W, float* __restrict__ outProc, int V)
{
  extern __shared__ __align__(16) char smem[];
  float* trg32 = (float*)smem;        // 16x256  original trg (residual + concat)
  float* t32   = trg32 + 4096;        // 16x256  post-LN1 t (residual 2)
  float* qctx  = t32 + 4096;          // 16x256  q, later ctx accumulator
  float* acc32 = qctx + 4096;         // 16x256  GEMM f32 scratch
  float* sc    = acc32 + 4096;        // 16x4 scores
  float* aw    = sc + 64;             // 16x4 softmax weights
  float* mu    = aw + 64;
  float* rsv   = mu + 16;
  _Float16* h16a = (_Float16*)(rsv + 16);   // 16x512 halves (A operand)
  _Float16* h16b = h16a + 8192;             // 16x512 halves (A operand)

  const int lane = threadIdx.x;
  const int v0 = blockIdx.x * 16;
  auto vidx = [&](int r) { int v = v0 + r; return (v < V) ? v : 0; };

  // trg = ori tile
  for (int idx = lane; idx < 4096; idx += 32) {
    int r = idx >> 8, c = idx & 255;
    float x = W.ori[(size_t)vidx(r) * 256 + c];
    trg32[idx] = x;
    h16a[r * 512 + c] = (_Float16)x;
  }
  __syncthreads();

  // q = trg @ Wq + bq
  wave_gemm<8>(h16a, 512, W.wq, W.bq, 16, nullptr, 0, qctx, 256, false, lane);
  __syncthreads();

  const float* srcs[4] = { pix, s1, s2, s3 };

  // scores_j = (q . k_j) / 16
  for (int j = 0; j < 4; ++j) {
    for (int idx = lane; idx < 4096; idx += 32) {
      int r = idx >> 8, c = idx & 255;
      h16b[r * 512 + c] = (_Float16)srcs[j][(size_t)vidx(r) * 256 + c];
    }
    __syncthreads();
    wave_gemm<8>(h16b, 512, W.wk, W.bk, 16, nullptr, 0, acc32, 256, false, lane);
    __syncthreads();
    {
      int r = lane & 15, part = lane >> 4;
      float s = 0.f; int cb = part * 128;
      for (int c = 0; c < 128; ++c) s += qctx[r * 256 + cb + c] * acc32[r * 256 + cb + c];
      s += __shfl_xor(s, 16, 32);
      if (lane < 16) sc[r * 4 + j] = s * 0.0625f;
    }
    __syncthreads();
  }

  // softmax over 4 sources
  if (lane < 16) {
    float m = sc[lane * 4];
    for (int j = 1; j < 4; ++j) m = fmaxf(m, sc[lane * 4 + j]);
    float e[4], ssum = 0.f;
    for (int j = 0; j < 4; ++j) { e[j] = __expf(sc[lane * 4 + j] - m); ssum += e[j]; }
    for (int j = 0; j < 4; ++j) aw[lane * 4 + j] = e[j] / ssum;
  }
  __syncthreads();

  // ctx = sum_j a_j * (src_j @ Wv + bv)   (reuse qctx)
  for (int idx = lane; idx < 4096; idx += 32) qctx[idx] = 0.f;
  __syncthreads();
  for (int j = 0; j < 4; ++j) {
    for (int idx = lane; idx < 4096; idx += 32) {
      int r = idx >> 8, c = idx & 255;
      h16b[r * 512 + c] = (_Float16)srcs[j][(size_t)vidx(r) * 256 + c];
    }
    __syncthreads();
    wave_gemm<8>(h16b, 512, W.wv, W.bv, 16, nullptr, 0, acc32, 256, false, lane);
    __syncthreads();
    for (int idx = lane; idx < 4096; idx += 32) {
      int r = idx >> 8;
      qctx[idx] += aw[r * 4 + j] * acc32[idx];
    }
    __syncthreads();
  }

  // attn_out = ctx @ Wo + bo ; t = LN1(trg + attn_out)
  for (int idx = lane; idx < 4096; idx += 32) {
    int r = idx >> 8, c = idx & 255;
    h16b[r * 512 + c] = (_Float16)qctx[idx];
  }
  __syncthreads();
  wave_gemm<8>(h16b, 512, W.wo, W.bo, 16, nullptr, 0, acc32, 256, false, lane);
  __syncthreads();
  for (int idx = lane; idx < 4096; idx += 32) acc32[idx] += trg32[idx];
  __syncthreads();
  ln_rows(acc32, mu, rsv, lane);
  __syncthreads();
  for (int idx = lane; idx < 4096; idx += 32) {
    int r = idx >> 8, c = idx & 255;
    float t = (acc32[idx] - mu[r]) * rsv[r] * ln1g[c] + ln1b[c];
    t32[idx] = t;
    h16a[r * 512 + c] = (_Float16)t;
  }
  __syncthreads();

  // FFN: ff = relu(t @ ff1) @ ff2 ; af = LN2(t + ff)
  wave_gemm<8>(h16a, 512, W.wf1, W.bf1, 32, h16b, 512, nullptr, 0, true,  lane);
  __syncthreads();
  wave_gemm<16>(h16b, 512, W.wf2, W.bf2, 16, nullptr, 0, acc32, 256, false, lane);
  __syncthreads();
  for (int idx = lane; idx < 4096; idx += 32) acc32[idx] += t32[idx];
  __syncthreads();
  ln_rows(acc32, mu, rsv, lane);
  __syncthreads();

  // cat = [trg | af]  (16 x 512)
  for (int idx = lane; idx < 4096; idx += 32) {
    int r = idx >> 8, c = idx & 255;
    h16a[r * 512 + c] = (_Float16)trg32[idx];
    float af = (acc32[idx] - mu[r]) * rsv[r] * ln2g[c] + ln2b[c];
    h16a[r * 512 + 256 + c] = (_Float16)af;
  }
  __syncthreads();

  // fusion head: relu(cat @ ca1) @ ca2  -> processed (16 x 128)
  wave_gemm<16>(h16a, 512, W.wc1, W.bc1, 16, h16b, 512, nullptr, 0, true,  lane);
  __syncthreads();
  wave_gemm<8>(h16b, 512, W.wc2, W.bc2, 8,  nullptr, 0, acc32, 256, false, lane);
  __syncthreads();
  for (int idx = lane; idx < 16 * 128; idx += 32) {
    int r = idx >> 7, c = idx & 127;
    int v = v0 + r;
    if (v < V) outProc[(size_t)v * 128 + c] = acc32[r * 256 + c];
  }
}

// ---------------------------------------------------------------------------
// Host launcher
// ---------------------------------------------------------------------------
extern "C" void kernel_launch(void* const* d_in, const int* in_sizes, int n_in,
                              void* d_out, int out_size, void* d_ws, size_t ws_size,
                              hipStream_t stream)
{
  const float* pt      = (const float*)d_in[0];
  const float* segfea  = (const float*)d_in[1];
  const float* pixfea  = (const float*)d_in[2];
  const int*   unq     = (const int*)d_in[3];
  const int*   unq_inv = (const int*)d_in[4];
  const int N = in_sizes[0] / 3;
  const int V = in_sizes[3] / 4;

  Params P;
  {
    int i = 5;
    P.bn0g=(const float*)d_in[i++]; P.bn0b=(const float*)d_in[i++];
    P.bn0m=(const float*)d_in[i++]; P.bn0v=(const float*)d_in[i++];
    P.l1W =(const float*)d_in[i++]; P.l1b =(const float*)d_in[i++];
    P.bn1g=(const float*)d_in[i++]; P.bn1b=(const float*)d_in[i++];
    P.bn1m=(const float*)d_in[i++]; P.bn1v=(const float*)d_in[i++];
    P.l2W =(const float*)d_in[i++]; P.l2b =(const float*)d_in[i++];
    P.bn2g=(const float*)d_in[i++]; P.bn2b=(const float*)d_in[i++];
    P.bn2m=(const float*)d_in[i++]; P.bn2v=(const float*)d_in[i++];
    P.l3W =(const float*)d_in[i++]; P.l3b =(const float*)d_in[i++];
    P.bn3g=(const float*)d_in[i++]; P.bn3b=(const float*)d_in[i++];
    P.bn3m=(const float*)d_in[i++]; P.bn3v=(const float*)d_in[i++];
    P.l4W =(const float*)d_in[i++]; P.l4b =(const float*)d_in[i++];
    P.qW  =(const float*)d_in[i++]; P.qb  =(const float*)d_in[i++];
    P.kW  =(const float*)d_in[i++]; P.kb  =(const float*)d_in[i++];
    P.vW  =(const float*)d_in[i++]; P.vb  =(const float*)d_in[i++];
    P.oW  =(const float*)d_in[i++]; P.ob  =(const float*)d_in[i++];
    P.ln1g=(const float*)d_in[i++]; P.ln1b=(const float*)d_in[i++];
    P.f1W =(const float*)d_in[i++]; P.f1b =(const float*)d_in[i++];
    P.f2W =(const float*)d_in[i++]; P.f2b =(const float*)d_in[i++];
    P.ln2g=(const float*)d_in[i++]; P.ln2b=(const float*)d_in[i++];
    P.c1W =(const float*)d_in[i++]; P.c1b =(const float*)d_in[i++];
    P.c2W =(const float*)d_in[i++]; P.c2b =(const float*)d_in[i++];
  }

  // Workspace carve
  char* ws = (char*)d_ws;
  size_t off = 0;
  auto alloc = [&](size_t bytes) -> void* {
    off = (off + 255) & ~(size_t)255;
    void* p = ws + off;
    off += bytes;
    return p;
  };
  WsPtrs W;
  W.ori = (float*)alloc((size_t)V * 256 * 4);
  W.a0 = (float*)alloc(4*4);   W.c0 = (float*)alloc(4*4);
  W.a1 = (float*)alloc(64*4);  W.a2 = (float*)alloc(128*4); W.a3 = (float*)alloc(256*4);
  W.b1 = (float*)alloc(64*4);  W.b2 = (float*)alloc(128*4);
  W.b3 = (float*)alloc(256*4); W.b4 = (float*)alloc(256*4);
  W.bq = (float*)alloc(256*4); W.bk = (float*)alloc(256*4);
  W.bv = (float*)alloc(256*4); W.bo = (float*)alloc(256*4);
  W.bf1= (float*)alloc(512*4); W.bf2= (float*)alloc(256*4);
  W.bc1= (float*)alloc(256*4); W.bc2= (float*)alloc(128*4);
  W.w1 = (_Float16*)alloc(2048*2);
  W.w2 = (_Float16*)alloc(8192*2);
  W.w3 = (_Float16*)alloc(32768*2);
  W.w4 = (_Float16*)alloc(65536*2);
  W.wq = (_Float16*)alloc(65536*2);
  W.wk = (_Float16*)alloc(65536*2);
  W.wv = (_Float16*)alloc(65536*2);
  W.wo = (_Float16*)alloc(65536*2);
  W.wf1= (_Float16*)alloc(131072*2);
  W.wf2= (_Float16*)alloc(131072*2);
  W.wc1= (_Float16*)alloc(131072*2);
  W.wc2= (_Float16*)alloc(32768*2);

  // Output carve: (unq, processed, s1, s2, s3, pix, unq_inv)
  float* outF = (float*)d_out;
  int*   outUnq  = (int*)outF;
  float* outProc = outF + (size_t)V * 4;
  float* outS1   = outProc + (size_t)V * 128;
  float* outS2   = outS1 + (size_t)V * 256;
  float* outS3   = outS2 + (size_t)V * 256;
  float* outPix  = outS3 + (size_t)V * 256;
  int*   outInv  = (int*)(outPix + (size_t)V * 256);

  // 1. Fold BN into weights/biases
  fold_kernel<<<1, 512, 0, stream>>>(P, W);

  // 2. Pack weights into WMMA B-fragment order (f16)
  pack_kernel<<<512, 256, 0, stream>>>(P.l1W,   3,  64, 1, W.a0,   W.a1,   W.w1);
  pack_kernel<<<512, 256, 0, stream>>>(P.l2W,  64, 128, 2, nullptr, W.a2,  W.w2);
  pack_kernel<<<512, 256, 0, stream>>>(P.l3W, 128, 256, 4, nullptr, W.a3,  W.w3);
  pack_kernel<<<512, 256, 0, stream>>>(P.l4W, 256, 256, 8, nullptr, nullptr, W.w4);
  pack_kernel<<<512, 256, 0, stream>>>(P.qW,  256, 256, 8, nullptr, nullptr, W.wq);
  pack_kernel<<<512, 256, 0, stream>>>(P.kW,  256, 256, 8, nullptr, nullptr, W.wk);
  pack_kernel<<<512, 256, 0, stream>>>(P.vW,  256, 256, 8, nullptr, nullptr, W.wv);
  pack_kernel<<<512, 256, 0, stream>>>(P.oW,  256, 256, 8, nullptr, nullptr, W.wo);
  pack_kernel<<<512, 256, 0, stream>>>(P.f1W, 256, 512, 8, nullptr, nullptr, W.wf1);
  pack_kernel<<<512, 256, 0, stream>>>(P.f2W, 512, 256,16, nullptr, nullptr, W.wf2);
  pack_kernel<<<512, 256, 0, stream>>>(P.c1W, 512, 256,16, nullptr, nullptr, W.wc1);
  pack_kernel<<<512, 256, 0, stream>>>(P.c2W, 256, 128, 8, nullptr, nullptr, W.wc2);

  // 3. Init pooled buffers to -inf; copy integer outputs
  {
    long long total = (long long)V * 256;
    int blocks = (int)((total + 255) / 256);
    init_pool_kernel<<<blocks, 256, 0, stream>>>(W.ori, outS1, outS2, outS3, outPix, total);
    int m = (V * 4 > N) ? V * 4 : N;
    copy_int_kernel<<<(m + 255) / 256, 256, 0, stream>>>(unq, unq_inv, outUnq, outInv, V * 4, N);
  }

  // 4. Point MLP with fused segment-max scatter (WMMA)
  {
    int tiles = (N + 15) / 16;
    int blocks = (tiles + 3) / 4;
    size_t shmem = 4 * (2 * 16 * 256) * sizeof(_Float16);  // 64 KB
    mlp_kernel<<<blocks, 128, shmem, stream>>>(pt, unq_inv, W, N);
  }

  // 5. Segment-max pooling for segfea/pixfea
  {
    long long total = (long long)N * 256;
    int blocks = (int)((total + 255) / 256);
    pool_kernel<<<blocks, 256, 0, stream>>>(segfea, pixfea, unq_inv,
                                            outS1, outS2, outS3, outPix, total);
  }

  // 6. Per-voxel attention + FFN + fusion head (WMMA)
  {
    int blocks = (V + 15) / 16;
    size_t shmem = (size_t)(4 * 4096 + 64 + 64 + 16 + 16) * sizeof(float)
                 + (size_t)2 * 8192 * sizeof(_Float16);     // ~97 KB
    attn_kernel<<<blocks, 32, shmem, stream>>>(outS1, outS2, outS3, outPix,
                                               P.ln1g, P.ln1b, P.ln2g, P.ln2b,
                                               W, outProc, V);
  }
}
